// PointNetFeaturePropagation_81870666596639
// MI455X (gfx1250) — matrix operand order repaired and checked by reading, not compiled, not run on hardware
//
#include <hip/hip_runtime.h>
#include <stdint.h>

#define BB   8
#define NN   16384
#define SS   1024
#define D1C  128
#define D2C  256
#define C1   256
#define C2   128
#define KCAT 384  // D1C + D2C

typedef __bf16 bf16_t;
typedef __bf16 v16bf __attribute__((ext_vector_type(16)));
typedef float  v8f   __attribute__((ext_vector_type(8)));
typedef unsigned int u32;
typedef u32 u32x4  __attribute__((ext_vector_type(4)));
typedef int i32x4v __attribute__((ext_vector_type(4)));
typedef int i32x8v __attribute__((ext_vector_type(8)));

union Frag16 { v16bf v; u32x4 q[2]; };

// LDS layout produced by TDM pad feature: insert 4 dwords after every 64 data dwords.
__device__ __forceinline__ u32 padByteOff(u32 d) {   // d = data dword index
  return (d << 2) + ((d >> 6) << 4);                 // 4*d + 16*(d/64); 16B aligned when d%4==0
}

// 2-D TDM load, bf16 elements, row-padded into LDS (pad_interval=64dw, pad_amount=4dw).
__device__ __forceinline__ void tdm_load_2d(const void* gptr, u32 lds_off,
                                            u32 width, u32 rows, u32 stride) {
  uint64_t ga = (uint64_t)(uintptr_t)gptr;
  u32x4 g0;
  g0.x = 1u;                                          // count=1, user descriptor
  g0.y = lds_off;                                     // LDS byte address
  g0.z = (u32)ga;                                     // global_addr[31:0]
  g0.w = (u32)((ga >> 32) & 0x1FFFFFFu) | (2u << 30); // global_addr[56:32] | type=2
  i32x8v g1;
  g1[0] = (int)((1u << 16)        // data_size = 2 bytes
              | (1u << 20)        // pad_enable
              | (5u << 22)        // pad_interval: 64 dwords
              | (3u << 25));      // pad_amount: 4 dwords
  g1[1] = (int)((width & 0xFFFFu) << 16);                                  // tensor_dim0[15:0]
  g1[2] = (int)(((width >> 16) & 0xFFFFu) | ((rows & 0xFFFFu) << 16));     // dim0 hi | dim1 lo
  g1[3] = (int)(((rows >> 16) & 0xFFFFu) | ((width & 0xFFFFu) << 16));     // dim1 hi | tile_dim0
  g1[4] = (int)(rows & 0xFFFFu);                                           // tile_dim1 (tile_dim2=0)
  g1[5] = (int)stride;                                                     // tensor_dim0_stride lo
  g1[6] = 0; g1[7] = 0;
  i32x4v z4 = {0, 0, 0, 0};
#if __has_include(<hip/amd_detail/amd_gfx1250_TDM.h>)
  i32x8v z8 = {0, 0, 0, 0, 0, 0, 0, 0};
  __builtin_amdgcn_tensor_load_to_lds(g0, g1, z4, z4, z8, 0);
#else
  __builtin_amdgcn_tensor_load_to_lds(g0, g1, z4, z4, 0);
#endif
}

// ---------------- K0: weights fp32 -> bf16 ----------------
__global__ __launch_bounds__(256) void k0_convert(const float* __restrict__ W1,
                                                  const float* __restrict__ W2,
                                                  bf16_t* __restrict__ w1b,
                                                  bf16_t* __restrict__ w2b) {
  int i = blockIdx.x * 256 + threadIdx.x;
  if (i < C1 * KCAT) {
    w1b[i] = (bf16_t)W1[i];
  } else {
    int j = i - C1 * KCAT;
    if (j < C2 * C1) w2b[j] = (bf16_t)W2[j];
  }
}

// ---------------- K1: three_nn + interpolate + concat -> feat[b][n][384] bf16 ----------------
__global__ __launch_bounds__(256) void k1_interp(const float* __restrict__ xyz1,
                                                 const float* __restrict__ xyz2,
                                                 const float* __restrict__ p1,
                                                 const float* __restrict__ p2,
                                                 bf16_t* __restrict__ feat) {
  __shared__ float  sx[SS * 3];       // 12 KB: xyz2 for this batch
  __shared__ bf16_t tile[256 * 66];   // 33 KB: [point][c-chunk] transpose buffer
  const int b  = blockIdx.y;
  const int n0 = blockIdx.x * 256;
  const int t  = threadIdx.x;
  for (int s = t; s < SS * 3; s += 256) sx[s] = xyz2[(size_t)b * SS * 3 + s];
  __syncthreads();

  const int n = n0 + t;
  const size_t pb = ((size_t)b * NN + n) * 3;
  const float px = xyz1[pb], py = xyz1[pb + 1], pz = xyz1[pb + 2];

  float d0 = 3e38f, d1 = 3e38f, d2v = 3e38f;
  int   i0 = 0, i1 = 0, i2 = 0;
  for (int s = 0; s < SS; ++s) {
    float dx = px - sx[3 * s], dy = py - sx[3 * s + 1], dz = pz - sx[3 * s + 2];
    float dd = dx * dx + dy * dy + dz * dz;
    if (dd < d0)       { d2v = d1; i2 = i1; d1 = d0; i1 = i0; d0 = dd; i0 = s; }
    else if (dd < d1)  { d2v = d1; i2 = i1; d1 = dd; i1 = s; }
    else if (dd < d2v) { d2v = dd; i2 = s; }
  }
  float r0 = 1.f / (d0 + 1e-8f), r1 = 1.f / (d1 + 1e-8f), r2 = 1.f / (d2v + 1e-8f);
  float rs = 1.f / (r0 + r1 + r2);
  float w0 = r0 * rs, w1 = r1 * rs, w2 = r2 * rs;

  const float* p2b = p2 + (size_t)b * D2C * SS;
  const float* p1b = p1 + (size_t)b * D1C * NN;

  for (int chunk = 0; chunk < 6; ++chunk) {
    __syncthreads();
    const int cbase = chunk * 64;
    if (chunk < 2) {
      for (int cc = 0; cc < 64; ++cc)
        tile[t * 66 + cc] = (bf16_t)p1b[(size_t)(cbase + cc) * NN + n];
    } else {
      const int c2 = cbase - D1C;
      for (int cc = 0; cc < 64; ++cc) {
        const float* row = p2b + (size_t)(c2 + cc) * SS;
        tile[t * 66 + cc] = (bf16_t)(w0 * row[i0] + w1 * row[i1] + w2 * row[i2]);
      }
    }
    __syncthreads();
    // coalesced write-out: each thread stores 2 bf16 (one dword)
    bf16_t* fb = feat + ((size_t)b * NN + n0) * KCAT + cbase;
    const int cc = (t & 31) << 1;
    for (int it = 0; it < 32; ++it) {
      const int row = (it << 3) + (t >> 5);
      *(u32*)(fb + (size_t)row * KCAT + cc) = *(const u32*)(tile + row * 66 + cc);
    }
  }
}

// ---------------- K2: GEMM1 (y1 = W1*feat + b1) + BN stats ----------------
// feat tile (128 pts x 384 c, bf16) TDM-staged to LDS; W1^T B-frags from global.
// Per wave: 2 M-tiles x 8 o-tiles -> each B-fragment feeds 2 WMMAs (1.25 b128 loads/WMMA).
__global__ __launch_bounds__(256) void k2_gemm1(const bf16_t* __restrict__ feat,
                                                const bf16_t* __restrict__ w1b,
                                                const float*  __restrict__ b1,
                                                bf16_t* __restrict__ y1,
                                                float* __restrict__ gsum,
                                                float* __restrict__ gsq) {
  extern __shared__ char smem[];                 // [0,104448): padded feat tile
  float* lsum = (float*)(smem + 104448);         // [256]
  float* lsq  = lsum + C1;                       // [256]
  const int b = blockIdx.y, n0 = blockIdx.x * 128, tid = threadIdx.x;
  lsum[tid] = 0.f; lsq[tid] = 0.f;
  if (tid < 32) {
    tdm_load_2d(feat + ((size_t)b * NN + n0) * KCAT, (u32)(uintptr_t)smem, KCAT, 128, KCAT);
    __builtin_amdgcn_s_wait_tensorcnt(0);
  }
  __syncthreads();

  const int wave = tid >> 5, lane = tid & 31;
  const int g = lane >> 4, m = lane & 15;
  const int nrow0 = (wave & 3) * 32;             // M-tile pair: points [nrow0, nrow0+32)
  const int nrow1 = nrow0 + 16;
  const int obase = (wave >> 2) * 128;           // o-half: 8 o-tiles

  v8f acc0[8], acc1[8];
#pragma unroll
  for (int tt = 0; tt < 8; ++tt)
#pragma unroll
    for (int r = 0; r < 8; ++r) { acc0[tt][r] = 0.f; acc1[tt][r] = 0.f; }

  for (int k = 0; k < KCAT; k += 32) {
    Frag16 a0, a1;
    const u32 kk = (u32)((k + g * 8) >> 1);
    const u32 d0 = (u32)(nrow0 + m) * 192 + kk;
    const u32 d1 = (u32)(nrow1 + m) * 192 + kk;
    a0.q[0] = *(const u32x4*)(smem + padByteOff(d0));
    a0.q[1] = *(const u32x4*)(smem + padByteOff(d0 + 8));
    a1.q[0] = *(const u32x4*)(smem + padByteOff(d1));
    a1.q[1] = *(const u32x4*)(smem + padByteOff(d1 + 8));
#pragma unroll
    for (int tt = 0; tt < 8; ++tt) {
      const int o = obase + tt * 16 + m;
      Frag16 bw;
      const u32x4* wp = (const u32x4*)(w1b + (size_t)o * KCAT + k + g * 8);
      bw.q[0] = wp[0]; bw.q[1] = wp[2];
      acc0[tt] = __builtin_amdgcn_wmma_f32_16x16x32_bf16(
          false, a0.v, false, bw.v, (short)0, acc0[tt], false, false);
      acc1[tt] = __builtin_amdgcn_wmma_f32_16x16x32_bf16(
          false, a1.v, false, bw.v, (short)0, acc1[tt], false, false);
    }
  }

#pragma unroll
  for (int tt = 0; tt < 8; ++tt) {
    const int o = obase + tt * 16 + m;
    const float bias = b1[o];
    float s = 0.f, s2 = 0.f;
    bf16_t* yb0 = y1 + ((size_t)b * NN + n0 + nrow0 + g * 8) * C1 + o;
    bf16_t* yb1 = y1 + ((size_t)b * NN + n0 + nrow1 + g * 8) * C1 + o;
#pragma unroll
    for (int r = 0; r < 8; ++r) {
      const float ya = acc0[tt][r] + bias;
      const float yc = acc1[tt][r] + bias;
      s += ya + yc; s2 += ya * ya + yc * yc;
      yb0[(size_t)r * C1] = (bf16_t)ya;
      yb1[(size_t)r * C1] = (bf16_t)yc;
    }
    atomicAdd(&lsum[o], s);
    atomicAdd(&lsq[o],  s2);
  }
  __syncthreads();
  atomicAdd(&gsum[tid], lsum[tid]);
  atomicAdd(&gsq[tid],  lsq[tid]);
}

// ---------------- K3: finalize BN1 scale/shift ----------------
__global__ void k3_fin1(const float* gsum, const float* gsq, const float* g1,
                        const float* be1, float* scale, float* shift) {
  const int o = threadIdx.x;  // 256
  const float cnt = (float)(BB * NN);
  const float mean = gsum[o] / cnt;
  const float var  = gsq[o] / cnt - mean * mean;
  const float a = g1[o] * rsqrtf(var + 1e-5f);
  scale[o] = a;
  shift[o] = be1[o] - mean * a;
}

// ---------------- K3b: BN1 + ReLU elementwise rewrite of y1 ----------------
__global__ __launch_bounds__(256) void k3b_bnrelu1(bf16_t* y1, const float* __restrict__ scale,
                                                   const float* __restrict__ shift) {
  const size_t total = (size_t)BB * NN * C1;
  for (size_t i = (size_t)blockIdx.x * 256 + threadIdx.x; i < total; i += (size_t)gridDim.x * 256) {
    const int o = (int)(i & (C1 - 1));
    const float z = scale[o] * (float)y1[i] + shift[o];
    y1[i] = (bf16_t)(z > 0.f ? z : 0.f);
  }
}

// ---------------- K4: GEMM2 (y2 = W2*z + b2) + stats; pre-BN y2 -> d_out ----------------
// Per wave: 2 M-tiles x 4 o-tiles (1.5 b128 loads/WMMA).
__global__ __launch_bounds__(256) void k4_gemm2(const bf16_t* __restrict__ z1,
                                                const bf16_t* __restrict__ w2b,
                                                const float*  __restrict__ b2,
                                                float* __restrict__ out,
                                                float* __restrict__ gsum,
                                                float* __restrict__ gsq) {
  extern __shared__ char smem[];                 // [0,69632): padded z1 tile, later [128][130] f32
  float* lsum = (float*)(smem + 69632);          // [128]
  float* lsq  = lsum + C2;                       // [128]
  const int b = blockIdx.y, n0 = blockIdx.x * 128, tid = threadIdx.x;
  if (tid < C2) { lsum[tid] = 0.f; lsq[tid] = 0.f; }
  if (tid < 32) {
    tdm_load_2d(z1 + ((size_t)b * NN + n0) * C1, (u32)(uintptr_t)smem, C1, 128, C1);
    __builtin_amdgcn_s_wait_tensorcnt(0);
  }
  __syncthreads();

  const int wave = tid >> 5, lane = tid & 31;
  const int g = lane >> 4, m = lane & 15;
  const int nrow0 = (wave & 3) * 32;
  const int nrow1 = nrow0 + 16;
  const int obase = (wave >> 2) * 64;            // o-half: 4 o-tiles

  v8f acc0[4], acc1[4];
#pragma unroll
  for (int tt = 0; tt < 4; ++tt)
#pragma unroll
    for (int r = 0; r < 8; ++r) { acc0[tt][r] = 0.f; acc1[tt][r] = 0.f; }

  for (int k = 0; k < C1; k += 32) {
    Frag16 a0, a1;
    const u32 kk = (u32)((k + g * 8) >> 1);
    const u32 d0 = (u32)(nrow0 + m) * 128 + kk;
    const u32 d1 = (u32)(nrow1 + m) * 128 + kk;
    a0.q[0] = *(const u32x4*)(smem + padByteOff(d0));
    a0.q[1] = *(const u32x4*)(smem + padByteOff(d0 + 8));
    a1.q[0] = *(const u32x4*)(smem + padByteOff(d1));
    a1.q[1] = *(const u32x4*)(smem + padByteOff(d1 + 8));
#pragma unroll
    for (int tt = 0; tt < 4; ++tt) {
      const int o = obase + tt * 16 + m;
      Frag16 bw;
      const u32x4* wp = (const u32x4*)(w2b + (size_t)o * C1 + k + g * 8);
      bw.q[0] = wp[0]; bw.q[1] = wp[2];
      acc0[tt] = __builtin_amdgcn_wmma_f32_16x16x32_bf16(
          false, a0.v, false, bw.v, (short)0, acc0[tt], false, false);
      acc1[tt] = __builtin_amdgcn_wmma_f32_16x16x32_bf16(
          false, a1.v, false, bw.v, (short)0, acc1[tt], false, false);
    }
  }
  __syncthreads();                               // all waves done reading staged tile
  float* dt = (float*)smem;                      // [128 n][130] transpose buffer
#pragma unroll
  for (int tt = 0; tt < 4; ++tt) {
    const int o = obase + tt * 16 + m;
    const float bias = b2[o];
    float s = 0.f, s2 = 0.f;
#pragma unroll
    for (int r = 0; r < 8; ++r) {
      const float ya = acc0[tt][r] + bias;
      const float yc = acc1[tt][r] + bias;
      s += ya + yc; s2 += ya * ya + yc * yc;
      dt[(nrow0 + r + g * 8) * 130 + o] = ya;
      dt[(nrow1 + r + g * 8) * 130 + o] = yc;
    }
    atomicAdd(&lsum[o], s);
    atomicAdd(&lsq[o],  s2);
  }
  __syncthreads();
  // coalesced fp32 stores: out[b][o][n0+nl]
  {
    const int nl = tid & 127, ob = tid >> 7;
    float* ob0 = out + (size_t)b * C2 * NN + n0;
    for (int it = 0; it < 64; ++it) {
      const int o = (it << 1) + ob;
      ob0[(size_t)o * NN + nl] = dt[nl * 130 + o];
    }
  }
  if (tid < C2) { atomicAdd(&gsum[tid], lsum[tid]); atomicAdd(&gsq[tid], lsq[tid]); }
}

// ---------------- K5: finalize BN2 scale/shift ----------------
__global__ void k5_fin2(const float* gsum, const float* gsq, const float* g2,
                        const float* be2, float* scale, float* shift) {
  const int o = threadIdx.x;  // 128
  const float cnt = (float)(BB * NN);
  const float mean = gsum[o] / cnt;
  const float var  = gsq[o] / cnt - mean * mean;
  const float a = g2[o] * rsqrtf(var + 1e-5f);
  scale[o] = a;
  shift[o] = be2[o] - mean * a;
}

// ---------------- K6: BN2 + ReLU in-place on d_out ----------------
__global__ __launch_bounds__(256) void k6_bnrelu2(float* out, const float* __restrict__ scale,
                                                  const float* __restrict__ shift) {
  const size_t total = (size_t)BB * C2 * NN;
  for (size_t i = (size_t)blockIdx.x * 256 + threadIdx.x; i < total; i += (size_t)gridDim.x * 256) {
    const int o = (int)((i >> 14) & (C2 - 1));
    const float z = scale[o] * out[i] + shift[o];
    out[i] = z > 0.f ? z : 0.f;
  }
}

extern "C" void kernel_launch(void* const* d_in, const int* in_sizes, int n_in,
                              void* d_out, int out_size, void* d_ws, size_t ws_size,
                              hipStream_t stream) {
  const float* xyz1 = (const float*)d_in[0];
  const float* xyz2 = (const float*)d_in[1];
  const float* p1   = (const float*)d_in[2];
  const float* p2   = (const float*)d_in[3];
  const float* W1   = (const float*)d_in[4];
  const float* b1   = (const float*)d_in[5];
  const float* g1   = (const float*)d_in[6];
  const float* be1  = (const float*)d_in[7];
  const float* W2   = (const float*)d_in[8];
  const float* b2   = (const float*)d_in[9];
  const float* g2   = (const float*)d_in[10];
  const float* be2  = (const float*)d_in[11];
  float* out = (float*)d_out;

  char* ws = (char*)d_ws;
  bf16_t* feat = (bf16_t*)(ws);                  // 100,663,296 B
  bf16_t* y1   = (bf16_t*)(ws + 100663296);      //  67,108,864 B
  bf16_t* w1b  = (bf16_t*)(ws + 167772160);      //     196,608 B
  bf16_t* w2b  = (bf16_t*)(ws + 167968768);      //      65,536 B
  float*  st   = (float*)(ws + 168034304);       //       6,144 B
  if (ws_size < (size_t)168040448) return;

  float* sum1 = st;        float* sq1 = st + 256;
  float* sc1  = st + 512;  float* sh1 = st + 768;
  float* sum2 = st + 1024; float* sq2 = st + 1152;
  float* sc2  = st + 1280; float* sh2 = st + 1408;

  hipMemsetAsync(st, 0, 1536 * sizeof(float), stream);

  k0_convert<<<512, 256, 0, stream>>>(W1, W2, w1b, w2b);
  k1_interp<<<dim3(64, 8), 256, 0, stream>>>(xyz1, xyz2, p1, p2, feat);
  k2_gemm1<<<dim3(128, 8), 256, 106496, stream>>>(feat, w1b, b1, y1, sum1, sq1);
  k3_fin1<<<1, 256, 0, stream>>>(sum1, sq1, g1, be1, sc1, sh1);
  k3b_bnrelu1<<<8192, 256, 0, stream>>>(y1, sc1, sh1);
  k4_gemm2<<<dim3(128, 8), 256, 70656, stream>>>(y1, w2b, b2, out, sum2, sq2);
  k5_fin2<<<1, 128, 0, stream>>>(sum2, sq2, g2, be2, sc2, sh2);
  k6_bnrelu2<<<8192, 256, 0, stream>>>(out, sc2, sh2);
}
